// Connectivity_Branch_56564719288608
// MI455X (gfx1250) — matrix operand order repaired and checked by reading, not compiled, not run on hardware
//
#include <hip/hip_runtime.h>

// ---------------------------------------------------------------------------
// MI455X (gfx1250) implementation.
//
// Roofline: dominated by reading W0 (1000 x 50000 f32 = 200 MB) ->
// ~8.6 us at 23.3 TB/s. W0 GEMV = f32 FMA, coalesced b128 loads with
// non-temporal hints (W0 ~ 200MB > 192MB L2; keep x/graph data resident).
// W1 (1000x1000) / W2 (100x1000) GEMVs run on v_wmma_f32_16x16x32_f16:
// one wave per 16-row tile, W converted f32->f16 on the fly, vector in
// column 0 of B, f32 accumulation in C/D; unguarded main K loop + guarded
// tail tile; row bounds handled by pointer clamping + discarding writes.
// Graph phase: edge-parallel scatter with L2 atomics, float segment-max via
// the signed/unsigned bit-order trick.
// ---------------------------------------------------------------------------

#define FIN 8

typedef __attribute__((ext_vector_type(16))) _Float16 v16h;
typedef __attribute__((ext_vector_type(8)))  float    v8f;
typedef __attribute__((ext_vector_type(4)))  float    f32x4;

__device__ __forceinline__ void atomic_max_float_bits(unsigned int* addr, float v) {
    int vi = __float_as_int(v);
    if (vi >= 0) atomicMax((int*)addr, vi);            // positive floats: int order == float order
    else         atomicMin(addr, (unsigned int)vi);    // negative floats: reversed in unsigned
}

// ---- zero / -inf initialize aggregation scratch (re-done every call) -------
__global__ void init_aggr(float* agg_sum, unsigned int* agg_max, int* cnt,
                          int n_feat_total, int n_nodes) {
    int i = blockIdx.x * blockDim.x + threadIdx.x;
    if (i < n_feat_total) {
        agg_sum[i] = 0.0f;
        agg_max[i] = 0xFF800000u;   // -inf bit pattern
    }
    if (i < n_nodes) cnt[i] = 0;
}

// ---- edge scatter: one thread per edge, b128 gathers of the X row ----------
__global__ void edge_scatter(const float* __restrict__ X,
                             const int* __restrict__ src,
                             const int* __restrict__ dst,
                             float* agg_sum, unsigned int* agg_max, int* cnt,
                             int E) {
    int e = blockIdx.x * blockDim.x + threadIdx.x;
    if (e >= E) return;
    int s = src[e];
    int d = dst[e];
    const f32x4* xs = (const f32x4*)(X + (size_t)s * FIN);   // 32B row, 16B aligned
    f32x4 x0 = xs[0];
    f32x4 x1 = xs[1];
    float xv[FIN] = {x0.x, x0.y, x0.z, x0.w, x1.x, x1.y, x1.z, x1.w};
    float*        as = agg_sum + (size_t)d * FIN;
    unsigned int* am = agg_max + (size_t)d * FIN;
#pragma unroll
    for (int f = 0; f < FIN; ++f) {
        atomicAdd(&as[f], xv[f]);
        atomic_max_float_bits(&am[f], xv[f]);
    }
    atomicAdd(&cnt[d], 1);
}

// ---- per-node SAGE epilogue: mean/max aggr -> two 8->1 linears -> embedding
__global__ void node_sage(const float* __restrict__ X,
                          const float* __restrict__ agg_sum,
                          const unsigned int* __restrict__ agg_max,
                          const int* __restrict__ cnt,
                          const float* __restrict__ llw, const float* __restrict__ llb,
                          const float* __restrict__ lrw,
                          const float* __restrict__ l1w, const float* __restrict__ l1b,
                          const float* __restrict__ r1w,
                          float* __restrict__ x_emb, int N) {
    int i = blockIdx.x * blockDim.x + threadIdx.x;
    if (i >= N) return;
    float c   = (float)cnt[i];
    float inv = 1.0f / fmaxf(c, 1.0f);
    bool  has = c > 0.0f;
    float hm = llb[0];
    float hx = l1b[0];
#pragma unroll
    for (int f = 0; f < FIN; ++f) {
        float sm = agg_sum[i * FIN + f] * inv;
        float mx = has ? __uint_as_float(agg_max[i * FIN + f]) : 0.0f;
        float xi = X[i * FIN + f];
        hm = fmaf(sm, llw[f], fmaf(xi, lrw[f], hm));
        hx = fmaf(mx, l1w[f], fmaf(xi, r1w[f], hx));
    }
    hm = fmaxf(hm, 0.0f);
    hx = fmaxf(hx, 0.0f);
    x_emb[i] = hm + hx;
}

// ---- big bandwidth-bound GEMV: h0 = relu(W0 @ x + b0), one block per row --
// W row loads are non-temporal (stream 200MB without evicting L2-resident x).
__global__ void gemv_f32_row(const float* __restrict__ W,
                             const float* __restrict__ x,
                             const float* __restrict__ bias,
                             float* __restrict__ out, int K) {
    const int row = blockIdx.x;
    const int tid = threadIdx.x;
    const f32x4* wr = (const f32x4*)(W + (size_t)row * K);
    const f32x4* xv = (const f32x4*)x;
    const int K4 = K >> 2;
    float acc = 0.0f;
    for (int i = tid; i < K4; i += blockDim.x) {
        f32x4 w = __builtin_nontemporal_load(&wr[i]);
        f32x4 v = xv[i];
        acc = fmaf(w.x, v.x, acc);
        acc = fmaf(w.y, v.y, acc);
        acc = fmaf(w.z, v.z, acc);
        acc = fmaf(w.w, v.w, acc);
    }
    // scalar tail (K % 4)
    int tail = K & 3;
    if (tid < tail) {
        int k = K4 * 4 + tid;
        acc = fmaf(W[(size_t)row * K + k], x[k], acc);
    }
    __shared__ float red[256];
    red[tid] = acc;
    __syncthreads();
    for (int s = blockDim.x >> 1; s > 0; s >>= 1) {
        if (tid < s) red[tid] += red[tid + s];
        __syncthreads();
    }
    if (tid == 0) {
        float v = red[0] + bias[row];
        out[row] = fmaxf(v, 0.0f);
    }
}

// ---- WMMA GEMV: out = relu(W @ v + bias), W is MxK row-major f32 ----------
// One wave (32 threads) per 16-row tile; K walked in 32-wide steps through
// v_wmma_f32_16x16x32_f16. Vector lives in column 0 of B; lanes 0/16 hold D
// column 0 per the 16x16 f32 C/D layout. Main loop is fully unguarded
// (b128 loads); one guarded tail tile handles K % 32.
__global__ void gemv_wmma_f16(const float* __restrict__ W,
                              const float* __restrict__ v,
                              const float* __restrict__ bias,
                              float* __restrict__ out, int M, int K) {
    const int lane    = threadIdx.x;        // blockDim.x == 32
    const int halfSel = lane >> 4;          // 0: lanes 0-15, 1: lanes 16-31
    const int r       = lane & 15;
    const int m0      = blockIdx.x * 16;
    const int m       = m0 + r;
    // clamp row pointer: loads always in-bounds; invalid rows never written
    const int mc      = m < M ? m : (M - 1);
    const float* wrow = W + (size_t)mc * K;
    const bool n0     = (r == 0);           // column 0 of B

    v8f c = {0.f, 0.f, 0.f, 0.f, 0.f, 0.f, 0.f, 0.f};
    const int KF = K & ~31;                 // full 32-wide tiles

    for (int kb = 0; kb < KF; kb += 32) {
        // A tile: lane<16 -> K in [kb,kb+8)+[kb+16,kb+24);
        //         lane>=16 -> K in [kb+8,kb+16)+[kb+24,kb+32). Same row both halves.
        const int k0 = kb + halfSel * 8;
        const int k1 = k0 + 16;
        f32x4 wa0 = *(const f32x4*)(wrow + k0);
        f32x4 wa1 = *(const f32x4*)(wrow + k0 + 4);
        f32x4 wb0 = *(const f32x4*)(wrow + k1);
        f32x4 wb1 = *(const f32x4*)(wrow + k1 + 4);
        v16h a, b;
        a[0]  = (_Float16)wa0.x; a[1]  = (_Float16)wa0.y; a[2]  = (_Float16)wa0.z; a[3]  = (_Float16)wa0.w;
        a[4]  = (_Float16)wa1.x; a[5]  = (_Float16)wa1.y; a[6]  = (_Float16)wa1.z; a[7]  = (_Float16)wa1.w;
        a[8]  = (_Float16)wb0.x; a[9]  = (_Float16)wb0.y; a[10] = (_Float16)wb0.z; a[11] = (_Float16)wb0.w;
        a[12] = (_Float16)wb1.x; a[13] = (_Float16)wb1.y; a[14] = (_Float16)wb1.z; a[15] = (_Float16)wb1.w;
        // B tile: N = lane&15, same K order as A (k0..k0+7, k1..k1+7).
        const float* vp = v + k0;
        f32x4 v0 = *(const f32x4*)(vp);
        f32x4 v1 = *(const f32x4*)(vp + 4);
        f32x4 v2 = *(const f32x4*)(vp + 16);
        f32x4 v3 = *(const f32x4*)(vp + 20);
        b[0]  = n0 ? (_Float16)v0.x : (_Float16)0.f;
        b[1]  = n0 ? (_Float16)v0.y : (_Float16)0.f;
        b[2]  = n0 ? (_Float16)v0.z : (_Float16)0.f;
        b[3]  = n0 ? (_Float16)v0.w : (_Float16)0.f;
        b[4]  = n0 ? (_Float16)v1.x : (_Float16)0.f;
        b[5]  = n0 ? (_Float16)v1.y : (_Float16)0.f;
        b[6]  = n0 ? (_Float16)v1.z : (_Float16)0.f;
        b[7]  = n0 ? (_Float16)v1.w : (_Float16)0.f;
        b[8]  = n0 ? (_Float16)v2.x : (_Float16)0.f;
        b[9]  = n0 ? (_Float16)v2.y : (_Float16)0.f;
        b[10] = n0 ? (_Float16)v2.z : (_Float16)0.f;
        b[11] = n0 ? (_Float16)v2.w : (_Float16)0.f;
        b[12] = n0 ? (_Float16)v3.x : (_Float16)0.f;
        b[13] = n0 ? (_Float16)v3.y : (_Float16)0.f;
        b[14] = n0 ? (_Float16)v3.z : (_Float16)0.f;
        b[15] = n0 ? (_Float16)v3.w : (_Float16)0.f;
        c = __builtin_amdgcn_wmma_f32_16x16x32_f16(
                false, a, false, b, (short)0, c, false, false);
    }

    if (KF < K) {                           // guarded tail tile (K % 32)
        const int kb = KF;
        const int k0 = kb + halfSel * 8;
        const int k1 = k0 + 16;
        v16h a, b;
#pragma unroll
        for (int i = 0; i < 8; ++i) {
            a[i]     = (k0 + i < K) ? (_Float16)wrow[k0 + i] : (_Float16)0.f;
            a[8 + i] = (k1 + i < K) ? (_Float16)wrow[k1 + i] : (_Float16)0.f;
        }
#pragma unroll
        for (int i = 0; i < 8; ++i) {
            b[i]     = (n0 && (k0 + i) < K) ? (_Float16)v[k0 + i] : (_Float16)0.f;
            b[8 + i] = (n0 && (k1 + i) < K) ? (_Float16)v[k1 + i] : (_Float16)0.f;
        }
        c = __builtin_amdgcn_wmma_f32_16x16x32_f16(
                false, a, false, b, (short)0, c, false, false);
    }

    // D column 0: lane 0 -> rows m0..m0+7, lane 16 -> rows m0+8..m0+15
    if (n0) {
#pragma unroll
        for (int vv = 0; vv < 8; ++vv) {
            int row = m0 + vv + 8 * halfSel;
            if (row < M) {
                float val = c[vv] + bias[row];
                out[row] = fmaxf(val, 0.0f);
            }
        }
    }
}

extern "C" void kernel_launch(void* const* d_in, const int* in_sizes, int n_in,
                              void* d_out, int out_size, void* d_ws, size_t ws_size,
                              hipStream_t stream) {
    const float* X       = (const float*)d_in[0];
    const int*   ei      = (const int*)d_in[1];     // [2,E] flattened: src then dst
    const float* lin_l_w = (const float*)d_in[2];
    const float* lin_l_b = (const float*)d_in[3];
    const float* lin_r_w = (const float*)d_in[4];
    const float* lin_l1w = (const float*)d_in[5];
    const float* lin_l1b = (const float*)d_in[6];
    const float* lin_r1w = (const float*)d_in[7];
    const float* W0      = (const float*)d_in[8];
    const float* b0      = (const float*)d_in[9];
    const float* W1      = (const float*)d_in[10];
    const float* b1      = (const float*)d_in[11];
    const float* W2      = (const float*)d_in[12];
    const float* b2      = (const float*)d_in[13];

    const int N   = in_sizes[0] / FIN;      // 50000
    const int E   = in_sizes[1] / 2;        // 1600000
    const int HID = in_sizes[9];            // 1000
    const int R   = in_sizes[13];           // 100

    const int* src = ei;
    const int* dst = ei + E;

    float* out_emb   = (float*)d_out;       // [N] embedding
    float* out_final = out_emb + N;         // [R] MLP output

    // workspace layout (all re-initialized or fully overwritten each call)
    float*        agg_sum = (float*)d_ws;                               // N*8 f32
    unsigned int* agg_max = (unsigned int*)(agg_sum + (size_t)N * FIN); // N*8
    int*          cnt     = (int*)(agg_max + (size_t)N * FIN);          // N
    float*        h0      = (float*)(cnt + N);                          // HID (padded)
    float*        h1      = h0 + ((HID + 31) & ~31);                    // HID

    const int BT = 256;

    // 1) init aggregation scratch
    {
        int total = N * FIN;
        init_aggr<<<(total + BT - 1) / BT, BT, 0, stream>>>(agg_sum, agg_max, cnt, total, N);
    }
    // 2) edge scatter (sum/max/count via L2 atomics)
    edge_scatter<<<(E + BT - 1) / BT, BT, 0, stream>>>(X, src, dst, agg_sum, agg_max, cnt, E);
    // 3) per-node SAGE epilogue -> embedding (first output)
    node_sage<<<(N + BT - 1) / BT, BT, 0, stream>>>(
        X, agg_sum, agg_max, cnt,
        lin_l_w, lin_l_b, lin_r_w, lin_l1w, lin_l1b, lin_r1w,
        out_emb, N);
    // 4) h0 = relu(W0 @ x + b0)   -- 200 MB NT stream, bandwidth-bound, exact f32
    gemv_f32_row<<<HID, BT, 0, stream>>>(W0, out_emb, b0, h0, N);
    // 5) h1 = relu(W1 @ h0 + b1)  -- WMMA f16, f32 accumulate
    gemv_wmma_f16<<<(HID + 15) / 16, 32, 0, stream>>>(W1, h0, b1, h1, HID, HID);
    // 6) out = relu(W2 @ h1 + b2) -- WMMA f16, f32 accumulate (second output)
    gemv_wmma_f16<<<(R + 15) / 16, 32, 0, stream>>>(W2, h1, b2, out_final, R, HID);
}
